// TransformerBlock_13804024890015
// MI455X (gfx1250) — compile-verified, matrix operand order checked by reading, not run on hardware
//
#include <hip/hip_runtime.h>
#include <math.h>
#include <stdint.h>

// ---------------------------------------------------------------------------
// Problem constants (B=2, S=2048, D=2048, H=16, HD=128, FF=4D=8192)
// ---------------------------------------------------------------------------
#define BB 2
#define SS 2048
#define DD 2048
#define HH 16
#define HD 128
#define FF 8192
#define MTOT (BB * SS)          // 4096 rows total

typedef __attribute__((ext_vector_type(16))) __bf16 v16bf;
typedef __attribute__((ext_vector_type(8)))  float  v8f;

__device__ __forceinline__ v8f zero8() {
  v8f z;
#pragma unroll
  for (int i = 0; i < 8; ++i) z[i] = 0.0f;
  return z;
}

// A/B fragment load for v_wmma_*_16x16x32_bf16.
// 16-bit operand layout (ISA 7.12.2): lanes 0-15 -> elems 0..7 = K[0..7],
// elems 8..15 = K[16..23]; lanes 16-31 -> K[8..15] / K[24..31].
__device__ __forceinline__ v16bf load_frag_row(const __bf16* rowp, int lane) {
  const int kh = (lane & 16) ? 8 : 0;
  v16bf f;
#pragma unroll
  for (int i = 0; i < 8; ++i) f[i] = rowp[kh + i];
#pragma unroll
  for (int i = 0; i < 8; ++i) f[8 + i] = rowp[16 + kh + i];
  return f;
}

__device__ __forceinline__ v8f wmma_bf16(v16bf a, v16bf b, v8f c) {
  return __builtin_amdgcn_wmma_f32_16x16x32_bf16(false, a, false, b, (short)0, c,
                                                 false, false);
}

// CDNA5 async global->LDS copy (no VGPR round-trip), tracked by ASYNCcnt.
__device__ __forceinline__ void async_load_b128(uint32_t lds_off, const void* gaddr) {
  asm volatile("global_load_async_to_lds_b128 %0, %1, off"
               :
               : "v"(lds_off), "v"((uint64_t)(uintptr_t)gaddr)
               : "memory");
}
__device__ __forceinline__ void wait_async0() {
  asm volatile("s_wait_asynccnt 0x0" ::: "memory");
}
__device__ __forceinline__ uint32_t lds_off_of(const void* p) {
  return (uint32_t)(uintptr_t)p;  // low 32 bits = LDS byte offset
}

// pack two bf16 into a dword for ds_store_b32 transpose scatters
union bf16pk {
  __bf16 h[2];
  uint32_t u;
};

// ---------------------------------------------------------------------------
// fp32 -> bf16 conversion (weights)
// ---------------------------------------------------------------------------
__global__ void cvt_f32_bf16_kernel(const float* __restrict__ in,
                                    __bf16* __restrict__ out, long n) {
  long i = ((long)blockIdx.x * blockDim.x + threadIdx.x) * 4;
  if (i + 3 < n) {
    float4 f = *(const float4*)(in + i);
    out[i + 0] = (__bf16)f.x;
    out[i + 1] = (__bf16)f.y;
    out[i + 2] = (__bf16)f.z;
    out[i + 3] = (__bf16)f.w;
  }
}

// ---------------------------------------------------------------------------
// LayerNorm: one block per row of D=2048, 256 threads, bf16 out
// ---------------------------------------------------------------------------
__global__ void layernorm_bf16_kernel(const float* __restrict__ x,
                                      const float* __restrict__ scale,
                                      const float* __restrict__ shift,
                                      __bf16* __restrict__ out) {
  __shared__ float2 red[256];
  const int row = blockIdx.x;
  const int tid = threadIdx.x;
  const float* xr = x + (size_t)row * DD;
  float v[8];
  float s = 0.f, s2 = 0.f;
#pragma unroll
  for (int i = 0; i < 8; ++i) {
    float t = xr[tid + i * 256];
    v[i] = t;
    s += t;
    s2 += t * t;
  }
  red[tid] = make_float2(s, s2);
  __syncthreads();
  for (int off = 128; off > 0; off >>= 1) {
    if (tid < off) {
      red[tid].x += red[tid + off].x;
      red[tid].y += red[tid + off].y;
    }
    __syncthreads();
  }
  const float mean = red[0].x * (1.0f / DD);
  const float var = red[0].y * (1.0f / DD) - mean * mean;
  const float rstd = rsqrtf(var + 1e-5f);
  __bf16* orow = out + (size_t)row * DD;
#pragma unroll
  for (int i = 0; i < 8; ++i) {
    int c = tid + i * 256;
    orow[c] = (__bf16)(scale[c] * (v[i] - mean) * rstd + shift[c]);
  }
}

// ---------------------------------------------------------------------------
// Tiled WMMA GEMM: C[M,N] = A[M,K] * B[K,N]  (A,B bf16 row-major, f32 accum)
// 128x128 block tile, 8 waves (2x4), each wave 64x32 (4x2 WMMA tiles).
// Double-buffered LDS: A slab streamed with global_load_async_to_lds_b128
// (ASYNCcnt) for slab t+1 while WMMAs run on slab t.
// ---------------------------------------------------------------------------
template <bool HAS_BIAS, bool GELU, bool HAS_RES, bool OUT_BF16>
__global__ void __launch_bounds__(256, 1)
gemm_bf16_kernel(const __bf16* __restrict__ A,
                 const __bf16* __restrict__ Bm,
                 const float* __restrict__ bias,
                 const float* __restrict__ res,
                 float* __restrict__ outf,
                 __bf16* __restrict__ outb,
                 int M, int N, int K) {
  __shared__ __bf16 As[2][128][32];   // 16 KB
  __shared__ __bf16 Bt[2][128][32];   // 16 KB, Bt[b][n][k] = B[k0+k][n0+n]
  const int tid = threadIdx.x;
  const int lane = tid & 31;
  const int wave = tid >> 5;
  const int wm = wave >> 2;        // 0..1  (64-row half)
  const int wn = wave & 3;         // 0..3  (32-col strip)
  const int m0 = blockIdx.y * 128;
  const int n0 = blockIdx.x * 128;
  const int coll = lane & 15;

  // per-thread A-tile slot: 16 contiguous bf16 (2 x b128)
  const int a_r = (tid * 16) >> 5;
  const int a_c = (tid * 16) & 31;
  const uint32_t a_lds0 = lds_off_of(&As[0][a_r][a_c]);
  const uint32_t a_lds1 = lds_off_of(&As[1][a_r][a_c]);
  const __bf16* a_src_base = A + (size_t)(m0 + a_r) * K + a_c;

  // per-thread B-tile slot: k-pair x 8 columns -> 8 x ds_store_b32
  const int b_kp = (tid >> 4) * 2;   // 0,2,...,30
  const int b_c0 = (tid & 15) * 8;   // 0..120

  auto stage = [&](int buf, int k0) {
    // A slab via async DMA to LDS
    const uint32_t al = buf ? a_lds1 : a_lds0;
    async_load_b128(al, a_src_base + k0);
    async_load_b128(al + 16, a_src_base + k0 + 8);
    // B slab transposed: read two k-rows, write packed dwords
    const __bf16* s0 = Bm + (size_t)(k0 + b_kp) * N + n0 + b_c0;
    const __bf16* s1 = s0 + N;
#pragma unroll
    for (int i = 0; i < 8; ++i) {
      bf16pk pk;
      pk.h[0] = s0[i];
      pk.h[1] = s1[i];
      *reinterpret_cast<uint32_t*>(&Bt[buf][b_c0 + i][b_kp]) = pk.u;
    }
    if (k0 + 32 < K) {  // stream following slab toward L2 (global_prefetch_b8)
      __builtin_prefetch(A + (size_t)(m0 + (tid >> 1)) * K + k0 + 32, 0, 3);
      __builtin_prefetch(Bm + (size_t)(k0 + 32 + (tid >> 3)) * N + n0, 0, 3);
    }
  };

  v8f acc[4][2];
#pragma unroll
  for (int i = 0; i < 4; ++i)
#pragma unroll
    for (int j = 0; j < 2; ++j) acc[i][j] = zero8();

  const int NK = K >> 5;
  stage(0, 0);
  for (int kt = 0; kt < NK; ++kt) {
    wait_async0();       // this wave's DMA for slab kt done
    __syncthreads();     // all waves: slab kt resident (syncthreads waits dscnt)
    if (kt + 1 < NK) stage((kt + 1) & 1, (kt + 1) * 32);  // overlap with WMMAs
    const int cur = kt & 1;

    v16bf afr[4];
#pragma unroll
    for (int i = 0; i < 4; ++i)
      afr[i] = load_frag_row(&As[cur][wm * 64 + i * 16 + coll][0], lane);
    v16bf bfr[2];
#pragma unroll
    for (int j = 0; j < 2; ++j)
      bfr[j] = load_frag_row(&Bt[cur][wn * 32 + j * 16 + coll][0], lane);
#pragma unroll
    for (int i = 0; i < 4; ++i)
#pragma unroll
      for (int j = 0; j < 2; ++j) acc[i][j] = wmma_bf16(afr[i], bfr[j], acc[i][j]);
  }

  // Epilogue. C layout: col = n0.. + (lane&15); row = base + r + 8*(lane>=16).
  const float GC = 0.6628559838f;  // tanh(sqrt(2/pi)) — "buggy" gelu constant
  const int rbase = (lane & 16) ? 8 : 0;
#pragma unroll
  for (int i = 0; i < 4; ++i)
#pragma unroll
    for (int j = 0; j < 2; ++j) {
      const int gm = m0 + wm * 64 + i * 16 + rbase;
      const int gn = n0 + wn * 32 + j * 16 + coll;
      float bv = HAS_BIAS ? bias[gn] : 0.0f;
#pragma unroll
      for (int r = 0; r < 8; ++r) {
        float val = acc[i][j][r] + bv;
        if (GELU) val = 0.5f * val * (1.0f + GC * (val + 0.044715f * val * val * val));
        size_t off = (size_t)(gm + r) * N + gn;
        if (HAS_RES) val += res[off];
        if (OUT_BF16) outb[off] = (__bf16)val;
        else          outf[off] = val;
      }
    }
}

// ---------------------------------------------------------------------------
// Flash attention, causal. Grid (S/128, H, B), 256 threads (8 waves).
// Each wave owns a 16-row Q strip; KV streamed in 64-key tiles.
// K tile via async DMA; softmax in place on the S accumulators (no spills).
// ---------------------------------------------------------------------------
__global__ void __launch_bounds__(256, 1)
flash_attn_kernel(const __bf16* __restrict__ q,
                  const __bf16* __restrict__ k,
                  const __bf16* __restrict__ v,
                  __bf16* __restrict__ ctx) {
  __shared__ __bf16 Ks[64][128];    // 16 KB  (kv-major) == B^T layout for Q*K^T
  __shared__ __bf16 Vt[128][64];    // 16 KB  Vt[hd][kv] == B^T layout for P*V
  __shared__ __bf16 Ps[8][16][64];  // 16 KB  per-wave P staging (C->A relayout)

  const int tid = threadIdx.x;
  const int lane = tid & 31;
  const int wave = tid >> 5;
  const int coll = lane & 15;
  const int qt = blockIdx.x, h = blockIdx.y, b = blockIdx.z;
  const int qr0 = qt * 128 + wave * 16;

  // per-thread K-tile slot: 32 contiguous bf16 (4 x b128)
  const int k_r = (tid * 32) >> 7;
  const int k_c = (tid * 32) & 127;
  const uint32_t k_lds = lds_off_of(&Ks[k_r][k_c]);
  const __bf16* k_src_base = k + ((size_t)b * SS + k_r) * DD + h * HD + k_c;

  // per-thread V-tile slot: kv-pair x 16 hd columns (packed b32 scatters)
  const int v_r = (tid >> 3) * 2;    // 0,2,...,62
  const int v_c0 = (tid & 7) * 16;   // 0..112
  const __bf16* v_src_base = v + ((size_t)b * SS + v_r) * DD + h * HD + v_c0;

  // Q fragments, K-dim = HD = 128 -> 4 chunks of 32
  v16bf qf[4];
  const __bf16* qbase = q + ((size_t)b * SS + qr0 + coll) * DD + h * HD;
#pragma unroll
  for (int c = 0; c < 4; ++c) qf[c] = load_frag_row(qbase + c * 32, lane);

  v8f oacc[8];
#pragma unroll
  for (int i = 0; i < 8; ++i) oacc[i] = zero8();
  float mrow[8], lrow[8];
#pragma unroll
  for (int r = 0; r < 8; ++r) { mrow[r] = -__builtin_inff(); lrow[r] = 0.0f; }

  const float scale = 0.08838834764831845f;  // 1/sqrt(HD)
  const int rbase = (lane & 16) ? 8 : 0;
  const int nkv = (qt + 1) * 2;              // 64-key tiles covering causal span

  for (int it = 0; it < nkv; ++it) {
    const int kv0 = it * 64;
    __syncthreads();
    // K tile: 64x128 row-major via async DMA (4 x b128 per thread)
    {
      const __bf16* src = k_src_base + (size_t)kv0 * DD;
      async_load_b128(k_lds, src);
      async_load_b128(k_lds + 16, src + 8);
      async_load_b128(k_lds + 32, src + 16);
      async_load_b128(k_lds + 48, src + 24);
    }
    {  // V tile transposed: two kv rows -> packed dword scatters
      const __bf16* s0 = v_src_base + (size_t)kv0 * DD;
      const __bf16* s1 = s0 + DD;
#pragma unroll
      for (int i = 0; i < 16; ++i) {
        bf16pk pk;
        pk.h[0] = s0[i];
        pk.h[1] = s1[i];
        *reinterpret_cast<uint32_t*>(&Vt[v_c0 + i][v_r]) = pk.u;
      }
    }
    wait_async0();
    __syncthreads();

    // S = Q * K^T  (16 x 64 per wave)
    v8f sacc[4];
#pragma unroll
    for (int nt = 0; nt < 4; ++nt) sacc[nt] = zero8();
#pragma unroll
    for (int nt = 0; nt < 4; ++nt)
#pragma unroll
      for (int c = 0; c < 4; ++c) {
        v16bf bf = load_frag_row(&Ks[nt * 16 + coll][c * 32], lane);
        sacc[nt] = wmma_bf16(qf[c], bf, sacc[nt]);
      }

    // online softmax, in place on sacc (row lives in one 16-lane half;
    // xor masks 1/2/4/8 stay inside the half)
#pragma unroll
    for (int r = 0; r < 8; ++r) {
      const int qrow = qr0 + r + rbase;
      float rmax = -__builtin_inff();
#pragma unroll
      for (int nt = 0; nt < 4; ++nt) {
        const int kvi = kv0 + nt * 16 + coll;
        float sv = sacc[nt][r] * scale;
        sv = (kvi <= qrow) ? sv : -__builtin_inff();
        sacc[nt][r] = sv;
        rmax = fmaxf(rmax, sv);
      }
      rmax = fmaxf(rmax, __shfl_xor(rmax, 1));
      rmax = fmaxf(rmax, __shfl_xor(rmax, 2));
      rmax = fmaxf(rmax, __shfl_xor(rmax, 4));
      rmax = fmaxf(rmax, __shfl_xor(rmax, 8));
      const float mnew = fmaxf(mrow[r], rmax);
      const float alpha = __expf(mrow[r] - mnew);
      float rsum = 0.0f;
#pragma unroll
      for (int nt = 0; nt < 4; ++nt) {
        const float p = __expf(sacc[nt][r] - mnew);
        Ps[wave][r + rbase][nt * 16 + coll] = (__bf16)p;
        rsum += p;
      }
      rsum += __shfl_xor(rsum, 1);
      rsum += __shfl_xor(rsum, 2);
      rsum += __shfl_xor(rsum, 4);
      rsum += __shfl_xor(rsum, 8);
      lrow[r] = lrow[r] * alpha + rsum;
      mrow[r] = mnew;
#pragma unroll
      for (int nt2 = 0; nt2 < 8; ++nt2) oacc[nt2][r] *= alpha;
    }

    // O += P * V   (P: 16x64 A-operand via LDS relayout; V^T as B-transposed)
#pragma unroll
    for (int c = 0; c < 2; ++c) {
      v16bf pa = load_frag_row(&Ps[wave][coll][c * 32], lane);
#pragma unroll
      for (int nt2 = 0; nt2 < 8; ++nt2) {
        v16bf vb = load_frag_row(&Vt[nt2 * 16 + coll][c * 32], lane);
        oacc[nt2] = wmma_bf16(pa, vb, oacc[nt2]);
      }
    }
  }

  // normalize and write ctx (bf16)
#pragma unroll
  for (int r = 0; r < 8; ++r) {
    const float inv = 1.0f / lrow[r];
    const int grow = qr0 + r + rbase;
#pragma unroll
    for (int nt2 = 0; nt2 < 8; ++nt2) {
      int hd = nt2 * 16 + coll;
      ctx[((size_t)b * SS + grow) * DD + h * HD + hd] = (__bf16)(oacc[nt2][r] * inv);
    }
  }
}

// ---------------------------------------------------------------------------
// Orchestration
// ---------------------------------------------------------------------------
extern "C" void kernel_launch(void* const* d_in, const int* in_sizes, int n_in,
                              void* d_out, int out_size, void* d_ws, size_t ws_size,
                              hipStream_t stream) {
  const float* x        = (const float*)d_in[0];
  const float* wq       = (const float*)d_in[1];
  const float* wk       = (const float*)d_in[2];
  const float* wv       = (const float*)d_in[3];
  const float* wo       = (const float*)d_in[4];
  const float* bo       = (const float*)d_in[5];
  const float* w1       = (const float*)d_in[6];
  const float* b1       = (const float*)d_in[7];
  const float* w2       = (const float*)d_in[8];
  const float* b2       = (const float*)d_in[9];
  const float* ln_scale = (const float*)d_in[10];
  const float* ln_shift = (const float*)d_in[11];
  float* out = (float*)d_out;

  // workspace layout (bytes)
  const size_t MD2 = (size_t)MTOT * DD * 2;   // bf16 M x D    = 16.78 MB
  const size_t DD2 = (size_t)DD * DD * 2;     // bf16 D x D    =  8.39 MB
  const size_t DF2 = (size_t)DD * FF * 2;     // bf16 D x 4D   = 33.55 MB
  const size_t MD4 = (size_t)MTOT * DD * 4;   // f32  M x D    = 33.55 MB
  const size_t MF2 = (size_t)MTOT * FF * 2;   // bf16 M x 4D   = 67.11 MB

  char* ws = (char*)d_ws;
  // Zone Z (MF2 bytes): {hln, q, k, v} early, reused entirely as FFN activation
  __bf16* hln  = (__bf16*)(ws);
  __bf16* qb   = (__bf16*)(ws + MD2);
  __bf16* kb   = (__bf16*)(ws + 2 * MD2);
  __bf16* vb   = (__bf16*)(ws + 3 * MD2);
  __bf16* actb = (__bf16*)(ws);               // aliases hln/q/k/v after attention
  size_t off = MF2;
  __bf16* wqb = (__bf16*)(ws + off); off += DD2;
  __bf16* wkb = (__bf16*)(ws + off); off += DD2;
  __bf16* wvb = (__bf16*)(ws + off); off += DD2;
  __bf16* wob = (__bf16*)(ws + off); off += DD2;
  __bf16* w1b = (__bf16*)(ws + off); off += DF2;
  __bf16* w2b = (__bf16*)(ws + off); off += DF2;
  __bf16* ctxb = (__bf16*)(ws + off); off += MD2;
  __bf16* h2b  = (__bf16*)(ws + off); off += MD2;
  float*  x1   = (float*)(ws + off); off += MD4;
  (void)ws_size; (void)in_sizes; (void)n_in; (void)out_size;

  const dim3 blk(256);
  // 1) weights -> bf16
  {
    long nDD = (long)DD * DD, nDF = (long)DD * FF;
    cvt_f32_bf16_kernel<<<dim3((unsigned)(nDD / 1024)), blk, 0, stream>>>(wq, wqb, nDD);
    cvt_f32_bf16_kernel<<<dim3((unsigned)(nDD / 1024)), blk, 0, stream>>>(wk, wkb, nDD);
    cvt_f32_bf16_kernel<<<dim3((unsigned)(nDD / 1024)), blk, 0, stream>>>(wv, wvb, nDD);
    cvt_f32_bf16_kernel<<<dim3((unsigned)(nDD / 1024)), blk, 0, stream>>>(wo, wob, nDD);
    cvt_f32_bf16_kernel<<<dim3((unsigned)(nDF / 1024)), blk, 0, stream>>>(w1, w1b, nDF);
    cvt_f32_bf16_kernel<<<dim3((unsigned)(nDF / 1024)), blk, 0, stream>>>(w2, w2b, nDF);
  }
  // 2) LN1
  layernorm_bf16_kernel<<<dim3(MTOT), blk, 0, stream>>>(x, ln_scale, ln_shift, hln);
  // 3) Q,K,V projections
  gemm_bf16_kernel<false, false, false, true>
      <<<dim3(DD / 128, MTOT / 128), blk, 0, stream>>>(hln, wqb, nullptr, nullptr,
                                                       nullptr, qb, MTOT, DD, DD);
  gemm_bf16_kernel<false, false, false, true>
      <<<dim3(DD / 128, MTOT / 128), blk, 0, stream>>>(hln, wkb, nullptr, nullptr,
                                                       nullptr, kb, MTOT, DD, DD);
  gemm_bf16_kernel<false, false, false, true>
      <<<dim3(DD / 128, MTOT / 128), blk, 0, stream>>>(hln, wvb, nullptr, nullptr,
                                                       nullptr, vb, MTOT, DD, DD);
  // 4) causal flash attention
  flash_attn_kernel<<<dim3(SS / 128, HH, BB), blk, 0, stream>>>(qb, kb, vb, ctxb);
  // 5) output projection + bo + residual(x) -> x1 (f32)
  gemm_bf16_kernel<true, false, true, false>
      <<<dim3(DD / 128, MTOT / 128), blk, 0, stream>>>(ctxb, wob, bo, x, x1,
                                                       nullptr, MTOT, DD, DD);
  // 6) LN2 (same scale/shift per reference)
  layernorm_bf16_kernel<<<dim3(MTOT), blk, 0, stream>>>(x1, ln_scale, ln_shift, h2b);
  // 7) FFN up + bias + buggy gelu -> actb (bf16)
  gemm_bf16_kernel<true, true, false, true>
      <<<dim3(FF / 128, MTOT / 128), blk, 0, stream>>>(h2b, w1b, b1, nullptr,
                                                       nullptr, actb, MTOT, FF, DD);
  // 8) FFN down + bias + residual(x1) -> out (f32)
  gemm_bf16_kernel<true, false, true, false>
      <<<dim3(DD / 128, MTOT / 128), blk, 0, stream>>>(actb, w2b, b2, x1, out,
                                                       nullptr, MTOT, DD, FF);
}
